// PerCNet_4818953306115
// MI455X (gfx1250) — compile-verified, matrix operand order checked by reading, not run on hardware
//
#include <hip/hip_runtime.h>

// ---------------------------------------------------------------------------
// Problem constants (from reference): N=25000 nodes, E=400000 edges, F=256.
// ---------------------------------------------------------------------------
#define NN     25000
#define EE     400000
#define FF     256
#define K1     768          // 3*F, first-layer reduction dim
#define EPS    1e-5f

typedef unsigned short u16;
typedef unsigned int   u32;

typedef __attribute__((ext_vector_type(16))) __bf16 v16bf;
typedef __attribute__((ext_vector_type(8)))  __bf16 v8bf;
typedef __attribute__((ext_vector_type(8)))  float  v8f;

// fp32 -> bf16 round-to-nearest-even
__device__ __forceinline__ u16 f2bf(float f) {
    u32 u = __float_as_uint(f);
    u32 r = ((u >> 16) & 1u) + 0x7FFFu;
    return (u16)((u + r) >> 16);
}
__device__ __forceinline__ float bf2f(u16 h) {
    return __uint_as_float(((u32)h) << 16);
}
__device__ __forceinline__ u32 pack2(float lo, float hi) {
    return ((u32)f2bf(hi) << 16) | (u32)f2bf(lo);
}

// ---------------------------------------------------------------------------
// Pack W[n][k] (row-major, n=out feature, k=reduction) into per-lane WMMA
// B-fragment layout for v_wmma_f32_16x16x32_bf16:
//   lane l: n = nt*16 + (l&15), khalf = l>>4
//   bf16 element j (0..15): k = kt*32 + khalf*8 + (j<8 ? j : j+8)
// Linear layout: ((kt*16 + nt)*32 + lane)*16 + j  -> one 32B load per frag.
// ---------------------------------------------------------------------------
__global__ void pack_b(const float* __restrict__ W, int Kdim,
                       u16* __restrict__ outp) {
    int idx = blockIdx.x * blockDim.x + threadIdx.x;
    int total = (Kdim / 32) * 16 * 32 * 16;
    if (idx >= total) return;
    int j    = idx & 15;
    int lane = (idx >> 4) & 31;
    int nt   = (idx >> 9) & 15;
    int kt   = idx >> 13;
    int n    = nt * 16 + (lane & 15);
    int k    = kt * 32 + (lane >> 4) * 8 + (j < 8 ? j : j + 8);
    outp[idx] = f2bf(W[(size_t)n * Kdim + k]);
}

// ---------------------------------------------------------------------------
// Zero the node accumulator and the four 256-float stat arrays.
// ---------------------------------------------------------------------------
__global__ void zero_ws(float* __restrict__ acc, float* __restrict__ stats) {
    size_t gid = (size_t)blockIdx.x * blockDim.x + threadIdx.x;
    if (gid < (size_t)NN * FF) acc[gid] = 0.f;
    if (gid < 4 * FF)          stats[gid] = 0.f;
}

// ---------------------------------------------------------------------------
// Edge pass 1: fused gather + 2-layer MLP (both branches) in bf16 WMMA.
// 128 threads = 4 waves; each wave owns a 16-edge tile.
// Dynamic LDS layout:
//   [0,2048)                     : block stat arrays s_sum[256], s_sq[256]
//   2048 + wave*32768            : feat  16 x 768 bf16  (24576 B)
//   2048 + wave*32768 + 24576    : htile 16 x 256 bf16  ( 8192 B)
// ---------------------------------------------------------------------------
__global__ void edge_pass1(const float* __restrict__ x,
                           const int*   __restrict__ ei,
                           const float* __restrict__ ea,
                           const u16* __restrict__ w1s, const u16* __restrict__ w2s,
                           const u16* __restrict__ w1m, const u16* __restrict__ w2m,
                           const float* __restrict__ b1s, const float* __restrict__ b2s,
                           const float* __restrict__ b1m, const float* __restrict__ b2m,
                           u16* __restrict__ h_ws, u16* __restrict__ m_ws,
                           float* __restrict__ sumE, float* __restrict__ sqE) {
    extern __shared__ char smem[];
    float* s_sum = (float*)smem;            // 256 f32
    float* s_sq  = s_sum + FF;              // 256 f32
    const int wave = threadIdx.x >> 5;
    const int lane = threadIdx.x & 31;
    u16* feat  = (u16*)(smem + 2048 + wave * 32768);   // [16][768]
    u16* htile = feat + 16 * K1;                       // [16][256]

    for (int i = threadIdx.x; i < 2 * FF; i += blockDim.x) s_sum[i] = 0.f;
    __syncthreads();

    const int e0 = (blockIdx.x * 4 + wave) * 16;

    // ---- cooperative gather: feat[m][0:256)=x[dst], [256:512)=x[src],
    //      [512:768)=edge_attr; 2 lanes per edge row, fp32->bf16 ----
    {
        const int m    = lane >> 1;
        const int half = lane & 1;
        const int e    = e0 + m;
        const int srcn = ei[e];          // edge_index[0] = source (x_j)
        const int dstn = ei[EE + e];     // edge_index[1] = dest   (x_i)
        const float* p0 = x  + (size_t)dstn * FF;
        const float* p1 = x  + (size_t)srcn * FF;
        const float* p2 = ea + (size_t)e    * FF;
        #pragma unroll
        for (int seg = 0; seg < 3; ++seg) {
            const float* p = (seg == 0) ? p0 : ((seg == 1) ? p1 : p2);
            for (int c = half * 4; c < FF; c += 8) {
                float4 v = *(const float4*)(p + c);
                u32 lo = pack2(v.x, v.y);
                u32 hi = pack2(v.z, v.w);
                *(uint2*)&feat[m * K1 + seg * FF + c] = make_uint2(lo, hi);
            }
        }
    }
    // feat/htile are per-wave: same-wave LDS ordering (DScnt) suffices.

    const int mrow  = lane & 15;
    const int khalf = lane >> 4;

    for (int branch = 0; branch < 2; ++branch) {
        const u16*   w1  = branch ? w1m : w1s;
        const u16*   w2  = branch ? w2m : w2s;
        const float* bb1 = branch ? b1m : b1s;
        const float* bb2 = branch ? b2m : b2s;
        u16*         ows = branch ? m_ws : h_ws;

        // ---------------- layer 1: [16x768] @ [768->256] + silu ------------
        for (int nh = 0; nh < 2; ++nh) {
            v8f c[8];
            #pragma unroll
            for (int t = 0; t < 8; ++t)
                c[t] = (v8f){0.f, 0.f, 0.f, 0.f, 0.f, 0.f, 0.f, 0.f};

            for (int kt = 0; kt < K1 / 32; ++kt) {
                // A fragment from LDS (two aligned 16B loads)
                v8bf lo = *(const v8bf*)&feat[mrow * K1 + kt * 32 + khalf * 8];
                v8bf hi = *(const v8bf*)&feat[mrow * K1 + kt * 32 + 16 + khalf * 8];
                v16bf a;
                #pragma unroll
                for (int i = 0; i < 8; ++i) { a[i] = lo[i]; a[i + 8] = hi[i]; }
                #pragma unroll
                for (int t = 0; t < 8; ++t) {
                    int nt = nh * 8 + t;
                    v16bf b = *(const v16bf*)(w1 + ((size_t)(kt * 16 + nt) * 32 + lane) * 16);
                    c[t] = __builtin_amdgcn_wmma_f32_16x16x32_bf16(
                               false, a, false, b, (short)0, c[t], false, false);
                }
            }
            // bias + silu, transpose-store to htile (A-major for layer 2)
            #pragma unroll
            for (int t = 0; t < 8; ++t) {
                const int n  = (nh * 8 + t) * 16 + mrow;   // output feature (col)
                const float bias = bb1[n];
                #pragma unroll
                for (int i = 0; i < 8; ++i) {
                    const int m = i + khalf * 8;           // edge row
                    float v = c[t][i] + bias;
                    float s = v / (1.f + __expf(-v));      // silu
                    htile[m * FF + n] = f2bf(s);
                }
            }
        }

        // ---------------- layer 2: [16x256] @ [256->256] -------------------
        for (int nh = 0; nh < 2; ++nh) {
            v8f c[8];
            #pragma unroll
            for (int t = 0; t < 8; ++t)
                c[t] = (v8f){0.f, 0.f, 0.f, 0.f, 0.f, 0.f, 0.f, 0.f};

            for (int kt = 0; kt < FF / 32; ++kt) {
                v8bf lo = *(const v8bf*)&htile[mrow * FF + kt * 32 + khalf * 8];
                v8bf hi = *(const v8bf*)&htile[mrow * FF + kt * 32 + 16 + khalf * 8];
                v16bf a;
                #pragma unroll
                for (int i = 0; i < 8; ++i) { a[i] = lo[i]; a[i + 8] = hi[i]; }
                #pragma unroll
                for (int t = 0; t < 8; ++t) {
                    int nt = nh * 8 + t;
                    v16bf b = *(const v16bf*)(w2 + ((size_t)(kt * 16 + nt) * 32 + lane) * 16);
                    c[t] = __builtin_amdgcn_wmma_f32_16x16x32_bf16(
                               false, a, false, b, (short)0, c[t], false, false);
                }
            }
            // bias + store bf16 to workspace; score branch also BN statistics
            #pragma unroll
            for (int t = 0; t < 8; ++t) {
                const int n  = (nh * 8 + t) * 16 + mrow;
                const float bias = bb2[n];
                float s = 0.f, q = 0.f;
                #pragma unroll
                for (int i = 0; i < 8; ++i) {
                    const int m = i + khalf * 8;
                    float v = c[t][i] + bias;
                    s += v; q += v * v;
                    ows[(size_t)(e0 + m) * FF + n] = f2bf(v);
                }
                if (branch == 0) {
                    atomicAdd(&s_sum[n], s);   // ds_add_f32
                    atomicAdd(&s_sq[n],  q);
                }
            }
        }
    }

    __syncthreads();
    for (int i = threadIdx.x; i < FF; i += blockDim.x) {
        atomicAdd(&sumE[i], s_sum[i]);
        atomicAdd(&sqE[i],  s_sq[i]);
    }
}

// ---------------------------------------------------------------------------
// Pass 2: msg = sigmoid(BN(h)) * m ; scatter-add into acc[dst].
// 64 threads per edge, 4 features each.
// ---------------------------------------------------------------------------
__global__ void msg_scatter(const u16* __restrict__ h_ws,
                            const u16* __restrict__ m_ws,
                            const int* __restrict__ ei,
                            const float* __restrict__ sumE,
                            const float* __restrict__ sqE,
                            const float* __restrict__ gamma,
                            const float* __restrict__ beta,
                            float* __restrict__ acc) {
    __shared__ float sc[FF], sh[FF];
    for (int i = threadIdx.x; i < FF; i += blockDim.x) {
        float mu  = sumE[i] * (1.f / EE);
        float var = sqE[i] * (1.f / EE) - mu * mu;
        float inv = rsqrtf(var + EPS);
        sc[i] = gamma[i] * inv;
        sh[i] = beta[i] - mu * gamma[i] * inv;
    }
    __syncthreads();

    size_t gid = (size_t)blockIdx.x * blockDim.x + threadIdx.x;
    int e = (int)(gid >> 6);
    int f = (int)(gid & 63) * 4;
    int dstn = ei[EE + e];
    uint2 hv = *(const uint2*)&h_ws[(size_t)e * FF + f];
    uint2 mv = *(const uint2*)&m_ws[(size_t)e * FF + f];
    u16 hu[4] = {(u16)hv.x, (u16)(hv.x >> 16), (u16)hv.y, (u16)(hv.y >> 16)};
    u16 mu4[4] = {(u16)mv.x, (u16)(mv.x >> 16), (u16)mv.y, (u16)(mv.y >> 16)};
    #pragma unroll
    for (int i = 0; i < 4; ++i) {
        float z = bf2f(hu[i]) * sc[f + i] + sh[f + i];
        float gate = 1.f / (1.f + __expf(-z));
        float msg = gate * bf2f(mu4[i]);
        atomicAdd(&acc[(size_t)dstn * FF + f + i], msg);   // global_atomic_add_f32
    }
}

// ---------------------------------------------------------------------------
// Node BN statistics over acc rows.
// ---------------------------------------------------------------------------
__global__ void node_stats(const float* __restrict__ acc,
                           float* __restrict__ sumN, float* __restrict__ sqN) {
    const int f  = threadIdx.x;       // 256 threads
    const int r0 = blockIdx.x * 32;
    float s = 0.f, q = 0.f;
    for (int r = r0; r < r0 + 32 && r < NN; ++r) {
        float v = acc[(size_t)r * FF + f];
        s += v; q += v * v;
    }
    atomicAdd(&sumN[f], s);
    atomicAdd(&sqN[f], q);
}

// ---------------------------------------------------------------------------
// Final: relu(x + BN(acc)).
// ---------------------------------------------------------------------------
__global__ void final_k(const float* __restrict__ x,
                        const float* __restrict__ acc,
                        const float* __restrict__ sumN,
                        const float* __restrict__ sqN,
                        const float* __restrict__ gamma,
                        const float* __restrict__ beta,
                        float* __restrict__ outp) {
    __shared__ float sc[FF], sh[FF];
    for (int i = threadIdx.x; i < FF; i += blockDim.x) {
        float mu  = sumN[i] * (1.f / NN);
        float var = sqN[i] * (1.f / NN) - mu * mu;
        float inv = rsqrtf(var + EPS);
        sc[i] = gamma[i] * inv;
        sh[i] = beta[i] - mu * gamma[i] * inv;
    }
    __syncthreads();
    size_t gid = (size_t)blockIdx.x * blockDim.x + threadIdx.x;
    int f = threadIdx.x;              // blockDim == 256 == FF
    float v = acc[gid] * sc[f] + sh[f];
    float r = x[gid] + v;
    outp[gid] = r > 0.f ? r : 0.f;
}

// ---------------------------------------------------------------------------
// Launch. Workspace layout (bytes):
//   0          w1s packed  (393216)       393216  w1m packed (393216)
//   786432     w2s packed  (131072)       917504  w2m packed (131072)
//   1048576    stats: sumE|sqE|sumN|sqN (4 x 1024)
//   1052672    acc   N*F f32   (25,600,000)
//   26652672   h_ws  E*F bf16  (204,800,000)
//   231452672  m_ws  E*F bf16  (204,800,000)   -> ~416 MB total
// ---------------------------------------------------------------------------
extern "C" void kernel_launch(void* const* d_in, const int* in_sizes, int n_in,
                              void* d_out, int out_size, void* d_ws, size_t ws_size,
                              hipStream_t stream) {
    const float* x        = (const float*)d_in[0];
    const int*   ei       = (const int*)  d_in[1];
    const float* ea       = (const float*)d_in[2];
    const float* w_full1  = (const float*)d_in[3];
    const float* b_full1  = (const float*)d_in[4];
    const float* w_full2  = (const float*)d_in[5];
    const float* b_full2  = (const float*)d_in[6];
    const float* w_n1     = (const float*)d_in[7];
    const float* b_n1     = (const float*)d_in[8];
    const float* w_n2     = (const float*)d_in[9];
    const float* b_n2     = (const float*)d_in[10];
    const float* gamma_i  = (const float*)d_in[11];
    const float* beta_i   = (const float*)d_in[12];
    const float* gamma_b  = (const float*)d_in[13];
    const float* beta_b   = (const float*)d_in[14];
    float* outp = (float*)d_out;

    char* ws = (char*)d_ws;
    u16*   w1s   = (u16*)  (ws + 0);
    u16*   w1m   = (u16*)  (ws + 393216);
    u16*   w2s   = (u16*)  (ws + 786432);
    u16*   w2m   = (u16*)  (ws + 917504);
    float* stats = (float*)(ws + 1048576);
    float* sumE  = stats;
    float* sqE   = stats + FF;
    float* sumN  = stats + 2 * FF;
    float* sqN   = stats + 3 * FF;
    float* acc   = (float*)(ws + 1052672);
    u16*   h_ws  = (u16*)  (ws + 26652672ull);
    u16*   m_ws  = (u16*)  (ws + 231452672ull);

    // 1) pack weights to bf16 WMMA B-fragment layout
    const int p1 = (K1 / 32) * 16 * 32 * 16;   // 196608 elems
    const int p2 = (FF / 32) * 16 * 32 * 16;   //  65536 elems
    pack_b<<<(p1 + 255) / 256, 256, 0, stream>>>(w_full1, K1, w1s);
    pack_b<<<(p1 + 255) / 256, 256, 0, stream>>>(w_n1,    K1, w1m);
    pack_b<<<(p2 + 255) / 256, 256, 0, stream>>>(w_full2, FF, w2s);
    pack_b<<<(p2 + 255) / 256, 256, 0, stream>>>(w_n2,    FF, w2m);

    // 2) zero accumulator + stat arrays (every call)
    zero_ws<<<(NN * FF + 255) / 256, 256, 0, stream>>>(acc, stats);

    // 3) WMMA edge pass: 6250 blocks x 128 threads, 130 KB dynamic LDS
    const size_t lds = 2048 + 4 * 32768;
    edge_pass1<<<EE / 64, 128, lds, stream>>>(
        x, ei, ea, w1s, w2s, w1m, w2m,
        b_full1, b_full2, b_n1, b_n2, h_ws, m_ws, sumE, sqE);

    // 4) gate * message, scatter-add (64 threads/edge)
    msg_scatter<<<(int)(((size_t)EE * 64) / 256), 256, 0, stream>>>(
        h_ws, m_ws, ei, sumE, sqE, gamma_i, beta_i, acc);

    // 5) node BN stats
    node_stats<<<(NN + 31) / 32, 256, 0, stream>>>(acc, sumN, sqN);

    // 6) relu(x + BN(acc))
    final_k<<<NN, 256, 0, stream>>>(x, acc, sumN, sqN, gamma_b, beta_b, outp);
}
